// Dualmodule_13374528160142
// MI455X (gfx1250) — matrix operand order repaired and checked by reading, not compiled, not run on hardware
//
#include <hip/hip_runtime.h>
#include <hip/hip_bf16.h>

// ---------------------------------------------------------------------------
// Types for CDNA5 WMMA (wave32): D(16x16 f32) = A(16x32 bf16) x B(32x16 bf16) + C
// ---------------------------------------------------------------------------
typedef __attribute__((ext_vector_type(16))) __bf16        bf16x16;
typedef __attribute__((ext_vector_type(8)))  float         f32x8;
typedef __attribute__((ext_vector_type(4)))  unsigned int  u32x4;

union FragB16 { u32x4 u[2]; bf16x16 v; };

__device__ __forceinline__ __bf16 f2bf(float f) {
  union { float f; unsigned int u; } in; in.f = f;
  unsigned int u = in.u;
  u += 0x7FFFu + ((u >> 16) & 1u);              // round-to-nearest-even
  union { unsigned short s; __bf16 b; } out; out.s = (unsigned short)(u >> 16);
  return out.b;
}

// A fragment: row-major A[M][K]; lane m = mt + (lane&15), half = lane>>4.
// elements e=0..7  <-> k = k0 + 8*half + e      (contiguous)
// elements e=8..15 <-> k = k0 + 16 + 8*half + (e-8) (contiguous)
__device__ __forceinline__ bf16x16 load_a_frag(const __bf16* __restrict__ A, int lda,
                                               int m0, int k0, int lane) {
  int m = m0 + (lane & 15);
  int half = lane >> 4;
  const __bf16* p = A + (size_t)m * lda + k0 + 8 * half;
  FragB16 f;
  f.u[0] = *reinterpret_cast<const u32x4*>(p);
  f.u[1] = *reinterpret_cast<const u32x4*>(p + 16);
  return f.v;
}

// B fragment: B is stored transposed, Bt[P][K]; lane col p = pt + (lane&15),
// elements e=0..7 <-> k = k0 + 8*half + e; e=8..15 <-> k = k0 + 16 + 8*half + (e-8)
__device__ __forceinline__ bf16x16 load_b_frag(const __bf16* __restrict__ Bt, int ldb,
                                               int p0, int k0, int lane) {
  int p = p0 + (lane & 15);
  int half = lane >> 4;
  const __bf16* q = Bt + (size_t)p * ldb + k0 + 8 * half;
  FragB16 f;
  f.u[0] = *reinterpret_cast<const u32x4*>(q);
  f.u[1] = *reinterpret_cast<const u32x4*>(q + 16);
  return f.v;
}

// ---------------------------------------------------------------------------
// Generic bf16 WMMA GEMM:  out[m][p] = epilogue( sum_k A[m][k] * Bt[p][k] )
// wave tile = 32(M) x 64(P) = 2x4 WMMA accumulators, block = (32,4) = 4 waves.
// grid = (P/64, M/128, nbatch*nsplit)
// Epilogue: v = v*scale[m]+bias[m]; v = gamma*v + res[m][p]; relu; then any of
//   f32 store (optionally atomicAdd, for split-K), bf16 store, transposed bf16 store.
// ---------------------------------------------------------------------------
__global__ __launch_bounds__(128) void k_gemm_wmma_bf16(
    const __bf16* __restrict__ A, int lda, size_t asb,
    const __bf16* __restrict__ Bt, int ldb, size_t bsb,
    int M, int P, int K, int kChunk, int nsplit,
    const float* __restrict__ scale, const float* __restrict__ bias,
    const float* __restrict__ gammap,
    const float* __restrict__ res, size_t rsb, int ldr,
    float* __restrict__ outF, size_t osb, int atomicF,
    __bf16* __restrict__ outB, size_t obsb,
    __bf16* __restrict__ outBT, int ldbt, int colOffT, size_t obtsb,
    int relu)
{
  const int lane = threadIdx.x;
  const int pt = blockIdx.x * 64;
  const int mt = (blockIdx.y * 4 + threadIdx.y) * 32;
  const int zz = blockIdx.z;
  const int batch = zz / nsplit;
  const int ks = zz - batch * nsplit;
  const int k0 = ks * kChunk;
  const int k1 = k0 + kChunk;

  A  += (size_t)batch * asb;
  Bt += (size_t)batch * bsb;

  f32x8 acc[2][4] = {};

  for (int k = k0; k < k1; k += 32) {
    // prefetch next K-slab of B into the cache hierarchy (global_prefetch_b8)
    __builtin_prefetch(Bt + (size_t)(pt + lane) * ldb + k + 32, 0, 0);

    bf16x16 a0 = load_a_frag(A, lda, mt,      k, lane);
    bf16x16 a1 = load_a_frag(A, lda, mt + 16, k, lane);
    bf16x16 b0 = load_b_frag(Bt, ldb, pt,      k, lane);
    bf16x16 b1 = load_b_frag(Bt, ldb, pt + 16, k, lane);
    bf16x16 b2 = load_b_frag(Bt, ldb, pt + 32, k, lane);
    bf16x16 b3 = load_b_frag(Bt, ldb, pt + 48, k, lane);

    acc[0][0] = __builtin_amdgcn_wmma_f32_16x16x32_bf16(false, a0, false, b0, (short)0, acc[0][0], false, false);
    acc[0][1] = __builtin_amdgcn_wmma_f32_16x16x32_bf16(false, a0, false, b1, (short)0, acc[0][1], false, false);
    acc[0][2] = __builtin_amdgcn_wmma_f32_16x16x32_bf16(false, a0, false, b2, (short)0, acc[0][2], false, false);
    acc[0][3] = __builtin_amdgcn_wmma_f32_16x16x32_bf16(false, a0, false, b3, (short)0, acc[0][3], false, false);
    acc[1][0] = __builtin_amdgcn_wmma_f32_16x16x32_bf16(false, a1, false, b0, (short)0, acc[1][0], false, false);
    acc[1][1] = __builtin_amdgcn_wmma_f32_16x16x32_bf16(false, a1, false, b1, (short)0, acc[1][1], false, false);
    acc[1][2] = __builtin_amdgcn_wmma_f32_16x16x32_bf16(false, a1, false, b2, (short)0, acc[1][2], false, false);
    acc[1][3] = __builtin_amdgcn_wmma_f32_16x16x32_bf16(false, a1, false, b3, (short)0, acc[1][3], false, false);
  }

  const float gsc = gammap ? *gammap : 1.0f;
  const int half = lane >> 4;
  const int nlo  = lane & 15;

  #pragma unroll
  for (int i = 0; i < 2; ++i) {
    #pragma unroll
    for (int j = 0; j < 4; ++j) {
      #pragma unroll
      for (int r = 0; r < 8; ++r) {
        // D layout: VGPR r -> row = 8*half + r ; col = lane&15
        int row = mt + i * 16 + half * 8 + r;
        int col = pt + j * 16 + nlo;
        float v = acc[i][j][r];
        if (scale) v = v * scale[row] + (bias ? bias[row] : 0.0f);
        if (res)        v = gsc * v + res[(size_t)batch * rsb + (size_t)row * ldr + col];
        else if (gammap) v = gsc * v;
        if (relu) v = fmaxf(v, 0.0f);
        if (outF) {
          float* o = outF + (size_t)batch * osb + (size_t)row * P + col;
          if (atomicF) unsafeAtomicAdd(o, v); else *o = v;
        }
        if (outB)  outB[(size_t)batch * obsb + (size_t)row * P + col] = f2bf(v);
        if (outBT) outBT[(size_t)batch * obtsb + (size_t)col * ldbt + colOffT + row] = f2bf(v);
      }
    }
  }
}

// ---------------------------------------------------------------------------
// f32 -> bf16 elementwise (weight conversion)
// ---------------------------------------------------------------------------
__global__ void k_cvt_bf16(const float* __restrict__ src, __bf16* __restrict__ dst, int n) {
  int i = blockIdx.x * blockDim.x + threadIdx.x;
  if (i < n) dst[i] = f2bf(src[i]);
}

// ---------------------------------------------------------------------------
// Tiled transpose: in f32 [batch][C][P] -> out bf16 [batch][P][C]
// grid (P/32, C/32, batch), block (32,8)
// ---------------------------------------------------------------------------
__global__ void k_transpose_bf16(const float* __restrict__ in, __bf16* __restrict__ out,
                                 int C, int P) {
  __shared__ float tile[32][33];
  int n  = blockIdx.z;
  int p0 = blockIdx.x * 32;
  int c0 = blockIdx.y * 32;
  const float* src = in  + (size_t)n * C * P;
  __bf16*      dst = out + (size_t)n * P * C;
  #pragma unroll
  for (int it = 0; it < 4; ++it) {
    int r = threadIdx.y + it * 8;
    tile[r][threadIdx.x] = src[(size_t)(c0 + r) * P + p0 + threadIdx.x];
  }
  __syncthreads();
  #pragma unroll
  for (int it = 0; it < 4; ++it) {
    int r = threadIdx.y + it * 8;
    dst[(size_t)(p0 + r) * C + c0 + threadIdx.x] = f2bf(tile[threadIdx.x][r]);
  }
}

// ---------------------------------------------------------------------------
// Depthwise 3x3 stride-2 conv + BN (small branch, memory bound)
// ---------------------------------------------------------------------------
__global__ void k_dwconv_bn(const float* __restrict__ in, const float* __restrict__ w,
                            const float* __restrict__ s, const float* __restrict__ b,
                            float* __restrict__ out, int C, int inH, int inW,
                            int outH, int outW, int total) {
  int idx = blockIdx.x * blockDim.x + threadIdx.x;
  if (idx >= total) return;
  int ox = idx % outW; int t = idx / outW;
  int oy = t % outH;   t /= outH;
  int c  = t % C;      int n = t / C;
  const float* ip = in + (size_t)(n * C + c) * inH * inW;
  const float* wp = w + c * 9;
  float acc = 0.f;
  #pragma unroll
  for (int dy = 0; dy < 3; ++dy) {
    int iy = 2 * oy + dy - 1;
    if ((unsigned)iy >= (unsigned)inH) continue;
    #pragma unroll
    for (int dx = 0; dx < 3; ++dx) {
      int ix = 2 * ox + dx - 1;
      if ((unsigned)ix >= (unsigned)inW) continue;
      acc += ip[iy * inW + ix] * wp[dy * 3 + dx];
    }
  }
  out[idx] = acc * s[c] + b[c];
}

// ---------------------------------------------------------------------------
// v (l3) f32 [4][512][256] -> v bf16 (same layout) and vT bf16 [4][256][512]
// ---------------------------------------------------------------------------
__global__ void k_make_v(const float* __restrict__ l3, __bf16* __restrict__ v,
                         __bf16* __restrict__ vT, int total) {
  int idx = blockIdx.x * blockDim.x + threadIdx.x;
  if (idx >= total) return;
  int p = idx & 255; int t = idx >> 8;
  int c = t & 511;   int n = t >> 9;
  float f = l3[idx];
  v[idx] = f2bf(f);
  vT[((size_t)n * 256 + p) * 512 + c] = f2bf(f);
}

// ---------------------------------------------------------------------------
// attn = softmax(-energy) row-wise; rows of 512, one block per row.
// (softmax(max_d(e)-e) == softmax(-e))
// ---------------------------------------------------------------------------
__global__ void k_softmax_neg(const float* __restrict__ e, __bf16* __restrict__ attn) {
  __shared__ float red[256];
  int row = blockIdx.x;
  const float* src = e + (size_t)row * 512;
  int t = threadIdx.x;
  float x0 = -src[t], x1 = -src[t + 256];
  red[t] = fmaxf(x0, x1);
  __syncthreads();
  for (int s = 128; s > 0; s >>= 1) { if (t < s) red[t] = fmaxf(red[t], red[t + s]); __syncthreads(); }
  float m = red[0];
  __syncthreads();
  float p0 = __expf(x0 - m), p1 = __expf(x1 - m);
  red[t] = p0 + p1;
  __syncthreads();
  for (int s = 128; s > 0; s >>= 1) { if (t < s) red[t] += red[t + s]; __syncthreads(); }
  float inv = 1.0f / red[0];
  attn[(size_t)row * 512 + t]       = f2bf(p0 * inv);
  attn[(size_t)row * 512 + t + 256] = f2bf(p1 * inv);
}

// ---------------------------------------------------------------------------
// Bilinear upsample (align_corners, 16->128) of lcam, fused:
//   slf = feat * local + feat, stored TRANSPOSED bf16 into catT[p][0:512]
// grid (P/32, C/32, batch), block (32,8), LDS-tiled transpose.
// ---------------------------------------------------------------------------
__global__ void k_upsample_fuse(const float* __restrict__ feat, const float* __restrict__ lcam,
                                __bf16* __restrict__ catT) {
  __shared__ float tile[32][33];
  const float S = 15.0f / 127.0f;
  int n  = blockIdx.z;
  int p0 = blockIdx.x * 32;
  int c0 = blockIdx.y * 32;
  #pragma unroll
  for (int it = 0; it < 4; ++it) {
    int r = threadIdx.y + it * 8;      // channel offset in tile
    int c = c0 + r;
    int p = p0 + threadIdx.x;
    int y = p >> 7, x = p & 127;
    float fy = y * S, fx = x * S;
    int y0 = (int)fy, x0 = (int)fx;
    float ty = fy - y0, tx = fx - x0;
    int y1 = min(y0 + 1, 15), x1 = min(x0 + 1, 15);
    const float* lc = lcam + ((size_t)n * 512 + c) * 256;
    float top = lc[y0 * 16 + x0] * (1.0f - tx) + lc[y0 * 16 + x1] * tx;
    float bot = lc[y1 * 16 + x0] * (1.0f - tx) + lc[y1 * 16 + x1] * tx;
    float local = top * (1.0f - ty) + bot * ty;
    float f = feat[((size_t)n * 512 + c) * 16384 + p];
    tile[r][threadIdx.x] = f * local + f;
  }
  __syncthreads();
  #pragma unroll
  for (int it = 0; it < 4; ++it) {
    int r = threadIdx.y + it * 8;      // pixel offset in tile
    int p = p0 + r;
    int c = c0 + threadIdx.x;
    catT[((size_t)n * 16384 + p) * 1024 + c] = f2bf(tile[threadIdx.x][r]);
  }
}

// ---------------------------------------------------------------------------
// Graph-branch small transforms (tiny FLOPs, scalar f32):
// t1[n][o<128][l<256] = bn_adj( sum_i w_adj[o][i] * z_idt[n][l][i] )
// ---------------------------------------------------------------------------
__global__ void k_adj(const float* __restrict__ zidt, const float* __restrict__ w_adj,
                      const float* __restrict__ s, const float* __restrict__ b,
                      float* __restrict__ t1, int total) {
  int idx = blockIdx.x * blockDim.x + threadIdx.x;
  if (idx >= total) return;
  int l = idx & 255; int t = idx >> 8;
  int o = t & 127;   int n = t >> 7;
  const float* zr = zidt + ((size_t)n * 256 + l) * 128;
  const float* wr = w_adj + o * 128;
  float acc = 0.f;
  for (int i = 0; i < 128; ++i) acc += wr[i] * zr[i];
  t1[((size_t)n * 128 + o) * 256 + l] = acc * s[o] + b[o];
}

// zb[n][o<256][d<128] = bf16( bn_wg( sum_c w_wg[o][c] * (t1[n][d][c] + z_idt[n][c][d]) ) )
__global__ void k_wg(const float* __restrict__ t1, const float* __restrict__ zidt,
                     const float* __restrict__ w_wg, const float* __restrict__ s,
                     const float* __restrict__ b, __bf16* __restrict__ zb, int total) {
  int idx = blockIdx.x * blockDim.x + threadIdx.x;
  if (idx >= total) return;
  int d = idx & 127; int t = idx >> 7;
  int o = t & 255;   int n = t >> 8;
  const float* tr = t1 + ((size_t)n * 128 + d) * 256;
  const float* zr = zidt + (size_t)n * 256 * 128 + d;
  const float* wr = w_wg + o * 256;
  float acc = 0.f;
  for (int c = 0; c < 256; ++c) acc += wr[c] * (tr[c] + zr[(size_t)c * 128]);
  zb[((size_t)n * 256 + o) * 128 + d] = f2bf(acc * s[o] + b[o]);
}

// ---------------------------------------------------------------------------
// Host-side orchestration
// ---------------------------------------------------------------------------
extern "C" void kernel_launch(void* const* d_in, const int* in_sizes, int n_in,
                              void* d_out, int out_size, void* d_ws, size_t ws_size,
                              hipStream_t stream) {
  (void)in_sizes; (void)n_in; (void)out_size; (void)ws_size;
  const float* feat     = (const float*)d_in[0];
  const float* w_phi    = (const float*)d_in[1];
  const float* bn_phi_s = (const float*)d_in[2];
  const float* bn_phi_b = (const float*)d_in[3];
  const float* w_theta  = (const float*)d_in[4];
  const float* bn_th_s  = (const float*)d_in[5];
  const float* bn_th_b  = (const float*)d_in[6];
  const float* w_adj    = (const float*)d_in[7];
  const float* bn_adj_s = (const float*)d_in[8];
  const float* bn_adj_b = (const float*)d_in[9];
  const float* w_wg     = (const float*)d_in[10];
  const float* bn_wg_s  = (const float*)d_in[11];
  const float* bn_wg_b  = (const float*)d_in[12];
  const float* w_conv3  = (const float*)d_in[13];
  const float* bn3_s    = (const float*)d_in[14];
  const float* bn3_b    = (const float*)d_in[15];
  const float* w_l1     = (const float*)d_in[16];
  const float* bnl1_s   = (const float*)d_in[17];
  const float* bnl1_b   = (const float*)d_in[18];
  const float* w_l2     = (const float*)d_in[19];
  const float* bnl2_s   = (const float*)d_in[20];
  const float* bnl2_b   = (const float*)d_in[21];
  const float* w_l3     = (const float*)d_in[22];
  const float* bnl3_s   = (const float*)d_in[23];
  const float* bnl3_b   = (const float*)d_in[24];
  const float* gamma    = (const float*)d_in[25];
  const float* w_final  = (const float*)d_in[26];
  const float* bnf_s    = (const float*)d_in[27];
  const float* bnf_b    = (const float*)d_in[28];
  float* out = (float*)d_out;

  // ---- workspace carve-up (256B aligned) ----
  char* ws = (char*)d_ws;
  size_t off = 0;
  auto take = [&](size_t bytes) -> void* {
    void* p = ws + off;
    off += (bytes + 255) & ~(size_t)255;
    return p;
  };
  __bf16* featT   = (__bf16*)take(4ull * 16384 * 512 * 2);   // [n][p][c]
  __bf16* wphi_b  = (__bf16*)take(256ull * 512 * 2);
  __bf16* wth_b   = (__bf16*)take(128ull * 512 * 2);
  __bf16* wc3_b   = (__bf16*)take(512ull * 256 * 2);
  __bf16* wfin_b  = (__bf16*)take(512ull * 1024 * 2);
  float*  l1      = (float*) take(4ull * 512 * 64 * 64 * 4);
  float*  l2      = (float*) take(4ull * 512 * 32 * 32 * 4);
  float*  l3      = (float*) take(4ull * 512 * 16 * 16 * 4);
  __bf16* v_b     = (__bf16*)take(4ull * 512 * 256 * 2);
  __bf16* vT_b    = (__bf16*)take(4ull * 256 * 512 * 2);
  float*  energy  = (float*) take(4ull * 512 * 512 * 4);
  __bf16* attn_b  = (__bf16*)take(4ull * 512 * 512 * 2);
  float*  lcam    = (float*) take(4ull * 512 * 256 * 4);
  __bf16* xsqz_b  = (__bf16*)take(4ull * 256 * 16384 * 2);
  __bf16* bm_b    = (__bf16*)take(4ull * 128 * 16384 * 2);
  __bf16* bmT_b   = (__bf16*)take(4ull * 16384 * 128 * 2);
  float*  zidt    = (float*) take(4ull * 256 * 128 * 4);
  float*  t1      = (float*) take(4ull * 128 * 256 * 4);
  __bf16* z_b     = (__bf16*)take(4ull * 256 * 128 * 2);
  __bf16* yT_b    = (__bf16*)take(4ull * 16384 * 256 * 2);
  __bf16* catT    = (__bf16*)take(4ull * 16384 * 1024 * 2); // [p][0:512]=slf, [512:1024]=g_out

  const dim3 gemmBlk(32, 4, 1);

  // 1) weight f32 -> bf16
  k_cvt_bf16<<<(256 * 512 + 255) / 256, 256, 0, stream>>>(w_phi, wphi_b, 256 * 512);
  k_cvt_bf16<<<(128 * 512 + 255) / 256, 256, 0, stream>>>(w_theta, wth_b, 128 * 512);
  k_cvt_bf16<<<(512 * 256 + 255) / 256, 256, 0, stream>>>(w_conv3, wc3_b, 512 * 256);
  k_cvt_bf16<<<(512 * 1024 + 255) / 256, 256, 0, stream>>>(w_final, wfin_b, 512 * 1024);

  // 2) feat -> featT (bf16, [p][c])
  k_transpose_bf16<<<dim3(16384 / 32, 512 / 32, 4), dim3(32, 8), 0, stream>>>(feat, featT, 512, 16384);

  // 3) local branch: three depthwise stride-2 convs + BN
  k_dwconv_bn<<<(4 * 512 * 64 * 64 + 255) / 256, 256, 0, stream>>>(feat, w_l1, bnl1_s, bnl1_b, l1, 512, 128, 128, 64, 64, 4 * 512 * 64 * 64);
  k_dwconv_bn<<<(4 * 512 * 32 * 32 + 255) / 256, 256, 0, stream>>>(l1, w_l2, bnl2_s, bnl2_b, l2, 512, 64, 64, 32, 32, 4 * 512 * 32 * 32);
  k_dwconv_bn<<<(4 * 512 * 16 * 16 + 255) / 256, 256, 0, stream>>>(l2, w_l3, bnl3_s, bnl3_b, l3, 512, 32, 32, 16, 16, 4 * 512 * 16 * 16);

  // 4) v / vT bf16
  k_make_v<<<(4 * 512 * 256 + 255) / 256, 256, 0, stream>>>(l3, v_b, vT_b, 4 * 512 * 256);

  // 5) energy = v @ v^T  (M=512,P=512,K=256) -> f32
  k_gemm_wmma_bf16<<<dim3(512 / 64, 512 / 128, 4), gemmBlk, 0, stream>>>(
      v_b, 256, 512ull * 256, v_b, 256, 512ull * 256,
      512, 512, 256, 256, 1,
      nullptr, nullptr, nullptr, nullptr, 0, 0,
      energy, 512ull * 512, 0, nullptr, 0, nullptr, 0, 0, 0, 0);

  // 6) attn = softmax(-energy) -> bf16
  k_softmax_neg<<<4 * 512, 256, 0, stream>>>(energy, attn_b);

  // 7) lcam = gamma * (attn @ v) + l3   (M=512,P=256,K=512)
  k_gemm_wmma_bf16<<<dim3(256 / 64, 512 / 128, 4), gemmBlk, 0, stream>>>(
      attn_b, 512, 512ull * 512, vT_b, 512, 256ull * 512,
      512, 256, 512, 512, 1,
      nullptr, nullptr, gamma, l3, 512ull * 256, 256,
      lcam, 512ull * 256, 0, nullptr, 0, nullptr, 0, 0, 0, 0);

  // 8) slf = feat * up(lcam) + feat -> catT[p][0:512] (transposed bf16)
  k_upsample_fuse<<<dim3(16384 / 32, 512 / 32, 4), dim3(32, 8), 0, stream>>>(feat, lcam, catT);

  // 9) x_sqz = BN(w_phi @ feat)  (M=256,P=16384,K=512) -> bf16
  k_gemm_wmma_bf16<<<dim3(16384 / 64, 256 / 128, 4), gemmBlk, 0, stream>>>(
      wphi_b, 512, 0, featT, 512, 16384ull * 512,
      256, 16384, 512, 512, 1,
      bn_phi_s, bn_phi_b, nullptr, nullptr, 0, 0,
      nullptr, 0, 0, xsqz_b, 256ull * 16384, nullptr, 0, 0, 0, 0);

  // 10) b_m = BN(w_theta @ feat) (M=128) -> bf16 normal + transposed
  k_gemm_wmma_bf16<<<dim3(16384 / 64, 128 / 128, 4), gemmBlk, 0, stream>>>(
      wth_b, 512, 0, featT, 512, 16384ull * 512,
      128, 16384, 512, 512, 1,
      bn_th_s, bn_th_b, nullptr, nullptr, 0, 0,
      nullptr, 0, 0, bm_b, 128ull * 16384, bmT_b, 128, 0, 16384ull * 128, 0);

  // 11) z_idt = x_sqz @ b_m^T (M=256,P=128,K=16384) split-K x8, f32 atomics
  hipMemsetAsync(zidt, 0, 4ull * 256 * 128 * 4, stream);
  k_gemm_wmma_bf16<<<dim3(128 / 64, 256 / 128, 4 * 8), gemmBlk, 0, stream>>>(
      xsqz_b, 16384, 256ull * 16384, bm_b, 16384, 128ull * 16384,
      256, 128, 16384, 2048, 8,
      nullptr, nullptr, nullptr, nullptr, 0, 0,
      zidt, 256ull * 128, 1, nullptr, 0, nullptr, 0, 0, 0, 0);

  // 12) graph transforms (tiny)
  k_adj<<<(4 * 128 * 256 + 255) / 256, 256, 0, stream>>>(zidt, w_adj, bn_adj_s, bn_adj_b, t1, 4 * 128 * 256);
  k_wg<<<(4 * 256 * 128 + 255) / 256, 256, 0, stream>>>(t1, zidt, w_wg, bn_wg_s, bn_wg_b, z_b, 4 * 256 * 128);

  // 13) y = z @ b_m (M=256,P=16384,K=128) -> yT bf16
  k_gemm_wmma_bf16<<<dim3(16384 / 64, 256 / 128, 4), gemmBlk, 0, stream>>>(
      z_b, 128, 256ull * 128, bmT_b, 128, 16384ull * 128,
      256, 16384, 128, 128, 1,
      nullptr, nullptr, nullptr, nullptr, 0, 0,
      nullptr, 0, 0, nullptr, 0, yT_b, 256, 0, 16384ull * 256, 0);

  // 14) g_out = relu(feat + BN(w_conv3 @ y)) -> catT[p][512:1024] (transposed bf16)
  k_gemm_wmma_bf16<<<dim3(16384 / 64, 512 / 128, 4), gemmBlk, 0, stream>>>(
      wc3_b, 256, 0, yT_b, 256, 16384ull * 256,
      512, 16384, 256, 256, 1,
      bn3_s, bn3_b, nullptr, feat, 512ull * 16384, 16384,
      nullptr, 0, 0, nullptr, 0, catT, 1024, 512, 16384ull * 1024, 1);

  // 15) out = BN(w_final @ concat(slf, g_out)) (M=512,P=16384,K=1024) -> f32
  k_gemm_wmma_bf16<<<dim3(16384 / 64, 512 / 128, 4), gemmBlk, 0, stream>>>(
      wfin_b, 1024, 0, catT, 1024, 16384ull * 1024,
      512, 16384, 1024, 1024, 1,
      bnf_s, bnf_b, nullptr, nullptr, 0, 0,
      out, 512ull * 16384, 0, nullptr, 0, nullptr, 0, 0, 0, 0);
}
